// NERModel_33749853012639
// MI455X (gfx1250) — compile-verified, matrix operand order checked by reading, not run on hardware
//
#include <hip/hip_runtime.h>
#include <hip/hip_bf16.h>
#include <math.h>

// ---------------- problem constants ----------------
#define B_   64
#define S_   512
#define W_   20
#define ECH  30
#define FCNN 30
#define HID  256
#define NG   1024          // 4*HID gates
#define T_   9
#define KP   512           // padded LSTM input width (490 -> 512)
#define ROWS (B_ * S_)     // 32768 token rows (time-major: row = t*B + b)

typedef __attribute__((ext_vector_type(16))) __bf16 v16bf;
typedef __attribute__((ext_vector_type(8)))  float  v8f;

__device__ __forceinline__ unsigned short f2bf(float f) {
    unsigned int x = __float_as_uint(f);
    x += 0x7FFFu + ((x >> 16) & 1u);           // round-to-nearest-even
    return (unsigned short)(x >> 16);
}
__device__ __forceinline__ float bf2f(unsigned short h) {
    return __uint_as_float(((unsigned int)h) << 16);
}
__device__ __forceinline__ float sigm(float x) { return 1.0f / (1.0f + __expf(-x)); }

union FragBF { v16bf v; float4 f[2]; };

__device__ __forceinline__ v16bf load_frag(const unsigned short* p) {
    FragBF fr;
    fr.f[0] = *reinterpret_cast<const float4*>(p);       // 8 bf16
    fr.f[1] = *reinterpret_cast<const float4*>(p + 8);   // 8 bf16
    return fr.v;
}

// async global -> LDS copy of 16 bytes (CDNA5, ASYNCcnt-tracked)
__device__ __forceinline__ void async_b128(unsigned lds_off, const void* gaddr) {
    asm volatile("global_load_async_to_lds_b128 %0, %1, off"
                 :: "v"(lds_off), "v"(gaddr) : "memory");
}

// ---------------- small prep kernels ----------------
__global__ void zero_u32(unsigned int* p, int n) {
    int i = blockIdx.x * blockDim.x + threadIdx.x;
    if (i < n) p[i] = 0u;
}

// convert fp32 (rows x srcK) -> bf16 (rows x dstK), zero-padding the K tail
__global__ void cvt_pad_bf16(const float* __restrict__ src, unsigned short* __restrict__ dst,
                             int rows, int srcK, int dstK) {
    int i = blockIdx.x * blockDim.x + threadIdx.x;
    int total = rows * dstK;
    if (i >= total) return;
    int r = i / dstK, k = i - r * dstK;
    dst[i] = (k < srcK) ? f2bf(src[r * srcK + k]) : (unsigned short)0;
}

__global__ void fuse_bias(const float* __restrict__ b1, const float* __restrict__ b2,
                          float* __restrict__ dst, int n) {
    int i = blockIdx.x * blockDim.x + threadIdx.x;
    if (i < n) dst[i] = b1[i] + b2[i];
}

// w_out (512 x 9, K-major) -> transposed/padded bf16 (64 x 512); b_out -> padded fp32 (64)
__global__ void cvt_wout(const float* __restrict__ w, const float* __restrict__ b,
                         unsigned short* __restrict__ wdst, float* __restrict__ bdst) {
    int i = blockIdx.x * blockDim.x + threadIdx.x;
    if (i >= 64 * KP) return;
    int t = i / KP, k = i - t * KP;
    wdst[i] = (t < T_) ? f2bf(w[k * T_ + t]) : (unsigned short)0;
    if (i < 64) bdst[i] = (i < T_) ? b[i] : 0.0f;
}

// ---------------- char-CNN + embedding concat -> X0 (bf16, padded to 512) ----------------
__global__ __launch_bounds__(128) void build_x0(
    const float* __restrict__ emb, const int* __restrict__ pos_ids,
    const int* __restrict__ dep_ids, const int* __restrict__ char_ids,
    const float* __restrict__ pos_tab, const float* __restrict__ dep_tab,
    const float* __restrict__ char_tab,
    const float* __restrict__ w2, const float* __restrict__ w3, const float* __restrict__ w4,
    const float* __restrict__ cb2, const float* __restrict__ cb3, const float* __restrict__ cb4,
    unsigned short* __restrict__ X)
{
    __shared__ float ce[W_][ECH];
    __shared__ float pooled[96];
    int tok = blockIdx.x;                 // tok = b*S + s
    int b = tok / S_, s = tok - b * S_;
    int tid = threadIdx.x;

    const int* cid = char_ids + (size_t)tok * W_;
    for (int idx = tid; idx < W_ * ECH; idx += blockDim.x) {
        int w = idx / ECH, e = idx - w * ECH;
        ce[w][e] = char_tab[cid[w] * ECH + e];
    }
    __syncthreads();

    if (tid < 3 * FCNN) {
        int kk = tid / FCNN;              // 0,1,2 -> kernel 2,3,4
        int f  = tid - kk * FCNN;
        int K  = kk + 2;
        const float* wgt  = (kk == 0) ? w2 : (kk == 1) ? w3 : w4;
        float bias = ((kk == 0) ? cb2 : (kk == 1) ? cb3 : cb4)[f];
        float best = 0.0f;                // relu then max => >= 0
        for (int p = 0; p <= W_ - K; ++p) {
            float acc = bias;
            for (int k = 0; k < K; ++k)
                for (int e = 0; e < ECH; ++e)
                    acc += ce[p + k][e] * wgt[(k * ECH + e) * FCNN + f];
            best = fmaxf(best, acc);
        }
        pooled[tid] = fmaxf(best, 0.0f);
    }
    __syncthreads();

    size_t row = (size_t)s * B_ + b;      // time-major row
    unsigned short* xr = X + row * KP;
    const float* er = emb + (size_t)tok * 300;
    int pid = pos_ids[tok], did = dep_ids[tok];
    for (int c = tid; c < KP; c += blockDim.x) {
        float v;
        if      (c < 300) v = er[c];
        else if (c < 350) v = pos_tab[pid * 50 + (c - 300)];
        else if (c < 400) v = dep_tab[did * 50 + (c - 350)];
        else if (c < 490) v = pooled[c - 400];
        else              v = 0.0f;
        xr[c] = f2bf(v);
    }
}

// ---------------- bf16 WMMA GEMM: C(MxN) = A(MxKp) * B(NxKp)^T + bias ----------------
// grid = (N/64, M/64), block = 128 (4 waves). Double-buffered async global->LDS staging.
template<bool BF16OUT>
__global__ __launch_bounds__(128) void gemm_bf16(
    const unsigned short* __restrict__ A, const unsigned short* __restrict__ Bm,
    const float* __restrict__ bias, void* __restrict__ Cout, int N, int Kp)
{
    __shared__ unsigned short lds_a[2][64][32];   // 2 x 4 KB
    __shared__ unsigned short lds_b[2][64][32];   // 2 x 4 KB
    int tid  = threadIdx.x;
    int lane = tid & 31;
    int wv   = tid >> 5;
    int l16  = lane & 15;
    int lh   = lane >> 4;
    int mBase = blockIdx.y * 64;
    int nBase = blockIdx.x * 64;
    int nk = Kp >> 5;

    auto issue_chunk = [&](int buf, int k0) {
        #pragma unroll
        for (int r = 0; r < 2; ++r) {
            int idx = tid + r * 128;            // 0..255 -> 64 rows x 4 x b128
            int row = idx >> 2;
            int col = (idx & 3) * 8;
            async_b128((unsigned)(uintptr_t)&lds_a[buf][row][col],
                       A + (size_t)(mBase + row) * Kp + k0 + col);
            async_b128((unsigned)(uintptr_t)&lds_b[buf][row][col],
                       Bm + (size_t)(nBase + row) * Kp + k0 + col);
        }
    };

    v8f acc[4] = {};
    issue_chunk(0, 0);                           // 4 async ops per wave
    for (int k = 0; k < nk; ++k) {
        int cur = k & 1;
        if (k + 1 < nk) {
            issue_chunk(cur ^ 1, (k + 1) << 5);  // prefetch next chunk
            asm volatile("s_wait_asynccnt 0x4" ::: "memory");  // current chunk landed
        } else {
            asm volatile("s_wait_asynccnt 0x0" ::: "memory");
        }
        __syncthreads();
        v16bf a = load_frag(&lds_a[cur][wv * 16 + l16][lh * 16]);
        #pragma unroll
        for (int tl = 0; tl < 4; ++tl) {
            v16bf bf = load_frag(&lds_b[cur][tl * 16 + l16][lh * 16]);
            acc[tl] = __builtin_amdgcn_wmma_f32_16x16x32_bf16(
                false, a, false, bf, (short)0, acc[tl], false, false);
        }
        __syncthreads();                          // buffer reuse fence
    }
    #pragma unroll
    for (int tl = 0; tl < 4; ++tl) {
        #pragma unroll
        for (int r = 0; r < 8; ++r) {
            int row = mBase + wv * 16 + r + lh * 8;      // WMMA C layout
            int col = nBase + tl * 16 + l16;
            float v = acc[tl][r] + bias[col];
            if (BF16OUT) reinterpret_cast<unsigned short*>(Cout)[(size_t)row * N + col] = f2bf(v);
            else         reinterpret_cast<float*>(Cout)[(size_t)row * N + col] = v;
        }
    }
}

// ---------------- one BiLSTM timestep (both directions via gridDim.z) ----------------
// grid = (4, 1, 2): 4 WGs each own a 64-wide hidden slice across ALL 4 gate blocks,
// so i/f/g/o for a given (b,j) land in the same lane -> fused cell epilogue.
// Whh + h_prev staged through LDS with async copies (removes 4x redundant wave loads).
__global__ __launch_bounds__(128) void lstm_step(
    const unsigned short* __restrict__ gates,   // [dir][S*B][1024] bf16 (X*Wih^T + bias)
    const unsigned short* __restrict__ whh,     // [dir][1024][256] bf16 (layer pre-offset)
    const unsigned short* __restrict__ hprev,   // [dir][64][256] bf16
    unsigned short* __restrict__ hnext,         // [dir][64][256] bf16 (double buffer)
    float* __restrict__ cstate,                 // [dir][64][256] fp32
    unsigned short* __restrict__ xout,          // [S*B][512] bf16 (layer output, concat fwd|bwd)
    int t)
{
    __shared__ unsigned short lds_w[2][256][32];   // 2 x 16 KB
    __shared__ unsigned short lds_h[2][64][32];    // 2 x 4 KB
    int dir  = blockIdx.z;
    int tdir = dir ? (S_ - 1 - t) : t;
    int j0   = blockIdx.x * 64;
    int tid  = threadIdx.x;
    int lane = tid & 31, wv = tid >> 5;
    int l16  = lane & 15, lh = lane >> 4;
    int mBase = wv * 16;                        // 4 waves cover B=64 batch rows

    const unsigned short* hp = hprev + (size_t)dir * B_ * HID;
    const unsigned short* wd = whh   + (size_t)dir * NG * HID;

    auto issue_chunk = [&](int buf, int k0) {
        #pragma unroll
        for (int r = 0; r < 8; ++r) {           // Whh: 256 rows x 4 x b128
            int idx = tid + r * 128;
            int row = idx >> 2;                 // 0..255 within slice
            int col = (idx & 3) * 8;
            int q = row >> 6, jr = row & 63;
            int nrow = q * 256 + j0 + jr;       // gate block q, hidden col j0+jr
            async_b128((unsigned)(uintptr_t)&lds_w[buf][row][col],
                       wd + (size_t)nrow * HID + k0 + col);
        }
        #pragma unroll
        for (int r = 0; r < 2; ++r) {           // h_prev: 64 rows x 4 x b128
            int idx = tid + r * 128;
            int row = idx >> 2;
            int col = (idx & 3) * 8;
            async_b128((unsigned)(uintptr_t)&lds_h[buf][row][col],
                       hp + (size_t)row * HID + k0 + col);
        }
    };

    v8f acc[4][4] = {};
    issue_chunk(0, 0);                           // 10 async ops per wave
    for (int k = 0; k < HID / 32; ++k) {
        int cur = k & 1;
        if (k + 1 < HID / 32) {
            issue_chunk(cur ^ 1, (k + 1) << 5);
            asm volatile("s_wait_asynccnt 0xa" ::: "memory");
        } else {
            asm volatile("s_wait_asynccnt 0x0" ::: "memory");
        }
        __syncthreads();
        v16bf a = load_frag(&lds_h[cur][mBase + l16][lh * 16]);
        #pragma unroll
        for (int q = 0; q < 4; ++q) {
            #pragma unroll
            for (int jt = 0; jt < 4; ++jt) {
                v16bf bf = load_frag(&lds_w[cur][q * 64 + jt * 16 + l16][lh * 16]);
                acc[q][jt] = __builtin_amdgcn_wmma_f32_16x16x32_bf16(
                    false, a, false, bf, (short)0, acc[q][jt], false, false);
            }
        }
        __syncthreads();
    }

    const unsigned short* gd = gates + (size_t)dir * ROWS * NG;
    float*          cd = cstate + (size_t)dir * B_ * HID;
    unsigned short* hd = hnext  + (size_t)dir * B_ * HID;

    #pragma unroll
    for (int jt = 0; jt < 4; ++jt) {
        #pragma unroll
        for (int r = 0; r < 8; ++r) {
            int brow = mBase + r + lh * 8;
            int j    = j0 + jt * 16 + l16;
            const unsigned short* grow = gd + ((size_t)tdir * B_ + brow) * NG;
            float gi = bf2f(grow[      j]) + acc[0][jt][r];
            float gf = bf2f(grow[256 + j]) + acc[1][jt][r];
            float gg = bf2f(grow[512 + j]) + acc[2][jt][r];
            float go = bf2f(grow[768 + j]) + acc[3][jt][r];
            float cp = cd[(size_t)brow * HID + j];
            float cn = sigm(gf) * cp + sigm(gi) * tanhf(gg);
            float hn = sigm(go) * tanhf(cn);
            cd[(size_t)brow * HID + j] = cn;
            unsigned short hb = f2bf(hn);
            hd[(size_t)brow * HID + j] = hb;
            xout[((size_t)tdir * B_ + brow) * KP + dir * HID + j] = hb;
        }
    }
}

// ---------------- CRF: numerator + forward algorithm + reduce to scalar loss ----------------
__global__ __launch_bounds__(64) void crf_kernel(
    const float* __restrict__ em,               // [S*B][64] fp32, cols 0..8 valid, time-major
    const int* __restrict__ tags,               // [B][S]
    const unsigned char* __restrict__ mask,     // [B][S] bool
    const float* __restrict__ start, const float* __restrict__ endv,
    const float* __restrict__ trans, float* __restrict__ out)
{
    __shared__ float red[64];
    int b = threadIdx.x;

    // numerator
    int tprev = tags[b * S_];
    float num = start[tprev] + em[(size_t)b * 64 + tprev];
    int lastTag = tprev;
    for (int t = 1; t < S_; ++t) {
        int tg = tags[b * S_ + t];
        float m = mask[b * S_ + t] ? 1.0f : 0.0f;
        num += m * (trans[tprev * T_ + tg] + em[((size_t)t * B_ + b) * 64 + tg]);
        if (mask[b * S_ + t]) lastTag = tg;
        tprev = tg;
    }
    num += endv[lastTag];

    // denominator: forward algorithm over T=9 states
    float alpha[T_];
    #pragma unroll
    for (int j = 0; j < T_; ++j) alpha[j] = start[j] + em[(size_t)b * 64 + j];
    for (int t = 1; t < S_; ++t) {
        const float* emt = em + ((size_t)t * B_ + b) * 64;
        float nxt[T_];
        #pragma unroll
        for (int j = 0; j < T_; ++j) {
            float mx = -1e30f;
            #pragma unroll
            for (int i = 0; i < T_; ++i) mx = fmaxf(mx, alpha[i] + trans[i * T_ + j]);
            float s = 0.0f;
            #pragma unroll
            for (int i = 0; i < T_; ++i) s += __expf(alpha[i] + trans[i * T_ + j] - mx);
            nxt[j] = mx + __logf(s) + emt[j];
        }
        bool m = mask[b * S_ + t] != 0;
        #pragma unroll
        for (int j = 0; j < T_; ++j) alpha[j] = m ? nxt[j] : alpha[j];
    }
    float mx = -1e30f;
    #pragma unroll
    for (int j = 0; j < T_; ++j) mx = fmaxf(mx, alpha[j] + endv[j]);
    float s = 0.0f;
    #pragma unroll
    for (int j = 0; j < T_; ++j) s += __expf(alpha[j] + endv[j] - mx);
    float den = mx + __logf(s);

    red[b] = num - den;
    __syncthreads();
    for (int st = 32; st > 0; st >>= 1) {
        if (b < st) red[b] += red[b + st];
        __syncthreads();
    }
    if (b == 0) out[0] = -red[0] / (float)B_;
}

// ---------------- host orchestration ----------------
extern "C" void kernel_launch(void* const* d_in, const int* in_sizes, int n_in,
                              void* d_out, int out_size, void* d_ws, size_t ws_size,
                              hipStream_t stream)
{
    (void)in_sizes; (void)n_in; (void)out_size; (void)ws_size;

    const float* embeddings = (const float*)d_in[0];
    const int*   pos_ids    = (const int*)d_in[1];
    const int*   dep_ids    = (const int*)d_in[2];
    const int*   char_ids   = (const int*)d_in[3];
    const unsigned char* attn_mask = (const unsigned char*)d_in[4];
    const int*   tags       = (const int*)d_in[5];
    const float* pos_table  = (const float*)d_in[6];
    const float* dep_table  = (const float*)d_in[7];
    const float* char_table = (const float*)d_in[8];
    const float* conv_w[3]  = {(const float*)d_in[9],  (const float*)d_in[10], (const float*)d_in[11]};
    const float* conv_b[3]  = {(const float*)d_in[12], (const float*)d_in[13], (const float*)d_in[14]};
    // lstm params: base 15, (layer*2 + dir)*4 + {Wih, Whh, bih, bhh}, dir order fwd,bwd
    const float* w_out      = (const float*)d_in[31];
    const float* b_out      = (const float*)d_in[32];
    const float* crf_start  = (const float*)d_in[33];
    const float* crf_end    = (const float*)d_in[34];
    const float* crf_trans  = (const float*)d_in[35];

    char* ws = (char*)d_ws;
    size_t off = 0;
    auto alloc = [&](size_t bytes) -> char* {
        char* p = ws + off;
        off = (off + bytes + 255) & ~(size_t)255;
        return p;
    };
    unsigned short* XA    = (unsigned short*)alloc((size_t)ROWS * KP * 2);        // 32 MB
    unsigned short* XB    = (unsigned short*)alloc((size_t)ROWS * KP * 2);        // 32 MB
    unsigned short* GATES = (unsigned short*)alloc((size_t)2 * ROWS * NG * 2);    // 128 MB
    unsigned short* WIH0  = (unsigned short*)alloc((size_t)2 * NG * KP * 2);
    unsigned short* WIH1  = (unsigned short*)alloc((size_t)2 * NG * KP * 2);
    unsigned short* WHH   = (unsigned short*)alloc((size_t)4 * NG * HID * 2);
    float*          BIAS  = (float*)alloc((size_t)4 * NG * 4);
    unsigned short* WOUT  = (unsigned short*)alloc((size_t)64 * KP * 2);
    float*          BOUTP = (float*)alloc((size_t)64 * 4);
    unsigned short* HA    = (unsigned short*)alloc((size_t)2 * B_ * HID * 2);     // 64 KB
    unsigned short* HB    = (unsigned short*)alloc((size_t)2 * B_ * HID * 2);     // 64 KB
    float*          CST   = (float*)alloc((size_t)2 * B_ * HID * 4);              // 128 KB
    float*          EMIS  = (float*)alloc((size_t)ROWS * 64 * 4);                 // 8 MB

    unsigned short* WIH[2] = {WIH0, WIH1};

    // --- weight prep (bf16 conversion, K-padding, bias fusion) ---
    for (int l = 0; l < 2; ++l) {
        for (int d = 0; d < 2; ++d) {
            int base = 15 + (l * 2 + d) * 4;
            const float* wih = (const float*)d_in[base + 0];
            const float* whh = (const float*)d_in[base + 1];
            const float* bih = (const float*)d_in[base + 2];
            const float* bhh = (const float*)d_in[base + 3];
            int srcK = (l == 0) ? 490 : 512;
            cvt_pad_bf16<<<(NG * KP + 255) / 256, 256, 0, stream>>>(
                wih, WIH[l] + (size_t)d * NG * KP, NG, srcK, KP);
            cvt_pad_bf16<<<(NG * HID + 255) / 256, 256, 0, stream>>>(
                whh, WHH + (size_t)(l * 2 + d) * NG * HID, NG, HID, HID);
            fuse_bias<<<(NG + 255) / 256, 256, 0, stream>>>(
                bih, bhh, BIAS + (size_t)(l * 2 + d) * NG, NG);
        }
    }
    cvt_wout<<<(64 * KP + 255) / 256, 256, 0, stream>>>(w_out, b_out, WOUT, BOUTP);

    // --- X0: char-CNN + concat + bf16 pack ---
    build_x0<<<ROWS, 128, 0, stream>>>(embeddings, pos_ids, dep_ids, char_ids,
                                       pos_table, dep_table, char_table,
                                       conv_w[0], conv_w[1], conv_w[2],
                                       conv_b[0], conv_b[1], conv_b[2], XA);

    // --- 2 BiLSTM layers ---
    unsigned short* xin = XA;
    unsigned short* xo  = XB;
    for (int l = 0; l < 2; ++l) {
        // hoisted input projections: gates[dir] = Xin * Wih^T + (bih+bhh)
        for (int d = 0; d < 2; ++d) {
            gemm_bf16<true><<<dim3(NG / 64, ROWS / 64), 128, 0, stream>>>(
                xin, WIH[l] + (size_t)d * NG * KP,
                BIAS + (size_t)(l * 2 + d) * NG,
                GATES + (size_t)d * ROWS * NG, NG, KP);
        }
        // zero h (both buffers) and c: contiguous region HA|HB|CST = 256 KB
        zero_u32<<<(65536 + 255) / 256, 256, 0, stream>>>((unsigned int*)HA, 65536);
        const unsigned short* whl = WHH + (size_t)l * 2 * NG * HID;
        for (int t = 0; t < S_; ++t) {
            unsigned short* hp = (t & 1) ? HB : HA;
            unsigned short* hn = (t & 1) ? HA : HB;
            lstm_step<<<dim3(4, 1, 2), 128, 0, stream>>>(
                GATES, whl, hp, hn, CST, xo, t);
        }
        unsigned short* tmp = xin; xin = xo; xo = tmp;
    }

    // --- emissions: Xfinal * w_out + b_out (N padded 9 -> 64) ---
    gemm_bf16<false><<<dim3(1, ROWS / 64), 128, 0, stream>>>(xin, WOUT, BOUTP, EMIS, 64, KP);

    // --- CRF loss ---
    crf_kernel<<<1, 64, 0, stream>>>(EMIS, tags, attn_mask, crf_start, crf_end, crf_trans,
                                     (float*)d_out);
}
